// IzhikevichSNNLayer_66614942761483
// MI455X (gfx1250) — matrix operand order repaired and checked by reading, not compiled, not run on hardware
//
#include <hip/hip_runtime.h>

typedef __attribute__((ext_vector_type(16))) _Float16 v16h;
typedef __attribute__((ext_vector_type(8)))  float    v8f;
typedef __attribute__((ext_vector_type(4)))  float    f32x4;

#define IN_F   4096
#define OUT_F  4096
#define BATCH  256
#define TSTEPS 8

// ---------------------------------------------------------------------------
// Kernel 1: raw = x @ W^T + b   via v_wmma_f32_16x16x32_f16
// Each wave computes a 16(M) x 64(N) tile: 1 A-fragment, 4 B-fragments,
// 4 f32 accumulators. K-loop steps by 32.
// Grid: (BATCH/16)*(OUT_F/64) = 1024 waves = 128 blocks of 256 threads.
// ---------------------------------------------------------------------------
__global__ __launch_bounds__(256) void gemm_wmma_kernel(
    const float* __restrict__ x,     // (BATCH, IN_F)
    const float* __restrict__ W,     // (OUT_F, IN_F) row-major -> W^T access is contiguous in K
    const float* __restrict__ bias,  // (OUT_F)
    float* __restrict__ raw)         // (BATCH, OUT_F)
{
    const int lane   = threadIdx.x & 31;
    const int wave   = threadIdx.x >> 5;
    const int waveId = blockIdx.x * 8 + wave;

    const int tilesN = OUT_F / 64;         // 64
    const int tm = waveId / tilesN;        // 0..15
    const int tn = waveId % tilesN;        // 0..63
    const int m0 = tm * 16;
    const int n0 = tn * 64;

    const int mrow = lane & 15;
    const int kgrp = lane >> 4;            // 0 or 1 (lane half)

    v8f c0 = {}, c1 = {}, c2 = {}, c3 = {};

    const float* arow = x + (size_t)(m0 + mrow) * IN_F;

    for (int k0 = 0; k0 < IN_F; k0 += 32) {
        // ---- A fragment: 16-bit A 16x32 layout ----
        // lane<16 : halves 0..7 -> K=k0+0..7,   halves 8..15 -> K=k0+16..23
        // lane>=16: halves 0..7 -> K=k0+8..15,  halves 8..15 -> K=k0+24..31
        const float* ap = arow + k0 + kgrp * 8;
        float af[16];
        *(f32x4*)(af + 0)  = *(const f32x4*)(ap + 0);
        *(f32x4*)(af + 4)  = *(const f32x4*)(ap + 4);
        *(f32x4*)(af + 8)  = *(const f32x4*)(ap + 16);
        *(f32x4*)(af + 12) = *(const f32x4*)(ap + 20);
        v16h a;
        #pragma unroll
        for (int i = 0; i < 16; ++i) a[i] = (_Float16)af[i];

        // ---- B fragments: B is KxN, lane holds column n; lane half h: elems 0..7 -> K 8h..8h+7, elems 8..15 -> K 16+8h..16+8h+7
        v16h bf[4];
        #pragma unroll
        for (int j = 0; j < 4; ++j) {
            const int n = n0 + j * 16 + (lane & 15);
            const float* bp = W + (size_t)n * IN_F + k0 + kgrp * 8;
            // prefetch next K-slab of W into caches (global_prefetch_b8)
            __builtin_prefetch(bp + 32, 0, 0);
            float bfl[16];
            *(f32x4*)(bfl + 0)  = *(const f32x4*)(bp + 0);
            *(f32x4*)(bfl + 4)  = *(const f32x4*)(bp + 4);
            *(f32x4*)(bfl + 8)  = *(const f32x4*)(bp + 16);
            *(f32x4*)(bfl + 12) = *(const f32x4*)(bp + 20);
            #pragma unroll
            for (int i = 0; i < 16; ++i) bf[j][i] = (_Float16)bfl[i];
        }

        c0 = __builtin_amdgcn_wmma_f32_16x16x32_f16(false, a, false, bf[0], (short)0, c0, false, false);
        c1 = __builtin_amdgcn_wmma_f32_16x16x32_f16(false, a, false, bf[1], (short)0, c1, false, false);
        c2 = __builtin_amdgcn_wmma_f32_16x16x32_f16(false, a, false, bf[2], (short)0, c2, false, false);
        c3 = __builtin_amdgcn_wmma_f32_16x16x32_f16(false, a, false, bf[3], (short)0, c3, false, false);
    }

    // ---- store: C/D layout: VGPR r, lane<16 -> M=r,N=lane ; lane>=16 -> M=8+r,N=lane-16
    const int rbase = (lane >> 4) * 8;
    const int ncol  = lane & 15;
    #pragma unroll
    for (int r = 0; r < 8; ++r) {
        const int row = m0 + rbase + r;
        float* orow = raw + (size_t)row * OUT_F + n0;
        orow[ 0 + ncol] = c0[r] + bias[n0 +  0 + ncol];
        orow[16 + ncol] = c1[r] + bias[n0 + 16 + ncol];
        orow[32 + ncol] = c2[r] + bias[n0 + 32 + ncol];
        orow[48 + ncol] = c3[r] + bias[n0 + 48 + ncol];
    }
}

// ---------------------------------------------------------------------------
// Kernel 2: fused LayerNorm -> tanh*|mag| -> 8-step Izhikevich dynamics.
// One 256-thread block per batch row; 16 features per thread (stride 256,
// coalesced). v/u/I state lives entirely in registers.
// ---------------------------------------------------------------------------
__global__ __launch_bounds__(256) void snn_fused_kernel(
    const float* __restrict__ raw,      // (BATCH, OUT_F)
    const float* __restrict__ ln_gamma, // (OUT_F)
    const float* __restrict__ ln_beta,  // (OUT_F)
    const float* __restrict__ mag,      // scalar
    const float* __restrict__ noise,    // (TSTEPS, BATCH, OUT_F)
    float* __restrict__ out)            // (BATCH, OUT_F)
{
    const int b   = blockIdx.x;
    const int tid = threadIdx.x;
    const float* row = raw + (size_t)b * OUT_F;

    float r[16];
    float s1 = 0.f, s2 = 0.f;
    #pragma unroll
    for (int j = 0; j < 16; ++j) {
        float v = row[tid + 256 * j];
        r[j] = v;
        s1 += v;
        s2 += v * v;
    }

    // wave32 reduction, then cross-wave via LDS
    #pragma unroll
    for (int off = 16; off > 0; off >>= 1) {
        s1 += __shfl_xor(s1, off, 32);
        s2 += __shfl_xor(s2, off, 32);
    }
    __shared__ float red1[8], red2[8];
    if ((tid & 31) == 0) { red1[tid >> 5] = s1; red2[tid >> 5] = s2; }
    __syncthreads();
    float t1 = 0.f, t2 = 0.f;
    #pragma unroll
    for (int w = 0; w < 8; ++w) { t1 += red1[w]; t2 += red2[w]; }

    const float mu   = t1 * (1.0f / OUT_F);
    const float var  = t2 * (1.0f / OUT_F) - mu * mu;
    const float rstd = rsqrtf(var + 1e-5f);
    const float cm   = fabsf(mag[0]);

    float vst[16], ust[16], Iin[16], acc[16];
    #pragma unroll
    for (int j = 0; j < 16; ++j) {
        const int n = tid + 256 * j;
        const float ln = (r[j] - mu) * rstd * ln_gamma[n] + ln_beta[n];
        Iin[j] = tanhf(ln) * cm;
        vst[j] = -65.0f;
        ust[j] = 0.2f * -65.0f;
        acc[j] = 0.0f;
    }

    for (int t = 0; t < TSTEPS; ++t) {
        const float* np = noise + ((size_t)t * BATCH + b) * OUT_F;
        #pragma unroll
        for (int j = 0; j < 16; ++j) {
            const int n = tid + 256 * j;
            const float It = Iin[j] + 0.3f * np[n];
            const float v  = vst[j];
            const float dv = (0.04f * v * v + 5.0f * v + 140.0f - ust[j] + It) * 0.5f;
            const float du = 0.02f * (0.2f * v - ust[j]) * 0.5f;
            const float vn = fminf(fmaxf(v + dv, -100.0f), 60.0f);
            const float un = fminf(fmaxf(ust[j] + du, -100.0f), 100.0f);
            const float s  = (vn >= 30.0f) ? 1.0f : 0.0f;
            vst[j] = vn * (1.0f - s) + (-65.0f) * s;
            ust[j] = un + 8.0f * s;
            acc[j] += s;
        }
    }

    float* orow = out + (size_t)b * OUT_F;
    #pragma unroll
    for (int j = 0; j < 16; ++j)
        orow[tid + 256 * j] = acc[j] * (1.0f / TSTEPS);
}

// ---------------------------------------------------------------------------
extern "C" void kernel_launch(void* const* d_in, const int* in_sizes, int n_in,
                              void* d_out, int out_size, void* d_ws, size_t ws_size,
                              hipStream_t stream) {
    const float* x        = (const float*)d_in[0];
    const float* W        = (const float*)d_in[1];
    const float* bias     = (const float*)d_in[2];
    const float* ln_gamma = (const float*)d_in[3];
    const float* ln_beta  = (const float*)d_in[4];
    const float* mag      = (const float*)d_in[5];
    const float* noise    = (const float*)d_in[6];
    float*       out      = (float*)d_out;

    // raw intermediate (4 MB). Prefer workspace; fall back to d_out (safe:
    // kernel 2 reads its row into registers and each thread re-writes exactly
    // the indices it read).
    float* raw = (ws_size >= (size_t)BATCH * OUT_F * sizeof(float))
                     ? (float*)d_ws : out;

    // 1024 waves: (BATCH/16) M-tiles x (OUT_F/64) N-tiles, 8 waves per block
    gemm_wmma_kernel<<<128, 256, 0, stream>>>(x, W, bias, raw);
    snn_fused_kernel<<<BATCH, 256, 0, stream>>>(raw, ln_gamma, ln_beta, mag, noise, out);
}
